// SoftMoEDynamics_18657337934736
// MI455X (gfx1250) — matrix operand order repaired and checked by reading, not compiled, not run on hardware
//
#include <hip/hip_runtime.h>
#include <hip/hip_bf16.h>

// ---------------------------------------------------------------------------
// SoftMoE dynamics, CDNA5 (gfx1250).
//   B=4096, N=64, LD=128, AD=32 -> D=160, E=16, S=1, H1=H2=256.
// Pipeline:
//   k_route   : x=concat(z,a); logits w = x@phi; dispatch softmax over N;
//               combine softmax over E; expert_in = dispatch^T @ x  (bf16,
//               stored expert-major so the MLP GEMM per expert is contiguous).
//   k_mlp     : fused 3-layer expert MLP, bf16 WMMA 16x16x32, all weights
//               resident in 320KB LDS (272KB bf16) -- weight-stationary.
//   k_combine : out[b,n,:] = sum_e combine_w[b,n,e] * eo[b,e,:].
// ---------------------------------------------------------------------------

#define BB   4096
#define NN   64
#define LDIM 128
#define ADIM 32
#define DDIM 160
#define EE   16
#define HH   256

typedef __attribute__((ext_vector_type(16))) __bf16       v16bf;
typedef __attribute__((ext_vector_type(8)))  float        v8f;
typedef __attribute__((ext_vector_type(4)))  unsigned int v4u;

__device__ inline __bf16 f2bf(float f) {
  union { float f; unsigned u; } v; v.f = f;
  unsigned r = v.u + 0x7FFFu + ((v.u >> 16) & 1u);  // round-to-nearest-even
  unsigned short h = (unsigned short)(r >> 16);
  return __builtin_bit_cast(__bf16, h);
}

union frag_u { v16bf v; v4u u[2]; };

// B fragment: 16 contiguous bf16 (32B) at p, 16B-aligned.
__device__ inline v16bf ld_frag_contig(const __bf16* p) {
  frag_u f;
  f.u[0] = *(const v4u*)p;
  f.u[1] = *(const v4u*)(p + 8);
  return f.v;
}
// A fragment: two 8-half chunks at p and p+16 (K-interleave per ISA layout).
__device__ inline v16bf ld_frag_a(const __bf16* p) {
  frag_u f;
  f.u[0] = *(const v4u*)p;
  f.u[1] = *(const v4u*)(p + 16);
  return f.v;
}

// ---------------------------------------------------------------------------
// Kernel 1: routing + dispatch mix.  One block per batch element b.
// ---------------------------------------------------------------------------
__global__ __launch_bounds__(256) void k_route(
    const float* __restrict__ z, const float* __restrict__ a,
    const float* __restrict__ phi,
    float* __restrict__ cw_out,          // [B][N][E] combine weights (f32)
    __bf16* __restrict__ xin_out) {      // [E][B][D] expert inputs (bf16)
  __shared__ float xs[NN * DDIM];        // 40 KB
  __shared__ float phis[DDIM * EE];      // 10 KB
  __shared__ float wlog[NN * EE];        //  4 KB
  __shared__ float dw[NN * EE];          //  4 KB

  const int b = blockIdx.x;
  const int t = threadIdx.x;

  for (int i = t; i < NN * LDIM; i += 256) {
    int n = i >> 7, d = i & (LDIM - 1);
    xs[n * DDIM + d] = z[(size_t)b * NN * LDIM + i];
  }
  for (int i = t; i < NN * ADIM; i += 256) {
    int n = i >> 5, d = i & (ADIM - 1);
    xs[n * DDIM + LDIM + d] = a[(size_t)b * NN * ADIM + i];
  }
  for (int i = t; i < DDIM * EE; i += 256) phis[i] = phi[i];
  __syncthreads();

  // logits: each thread does one n, four e's
  {
    int n = t & 63, e0 = (t >> 6) << 2;
    float a0 = 0.f, a1 = 0.f, a2 = 0.f, a3 = 0.f;
    for (int d = 0; d < DDIM; ++d) {
      float xv = xs[n * DDIM + d];
      a0 += xv * phis[d * EE + e0 + 0];
      a1 += xv * phis[d * EE + e0 + 1];
      a2 += xv * phis[d * EE + e0 + 2];
      a3 += xv * phis[d * EE + e0 + 3];
    }
    wlog[n * EE + e0 + 0] = a0; wlog[n * EE + e0 + 1] = a1;
    wlog[n * EE + e0 + 2] = a2; wlog[n * EE + e0 + 3] = a3;
  }
  __syncthreads();

  // dispatch softmax over N (per expert column), threads 0..15
  if (t < EE) {
    int e = t;
    float m = -1e30f;
    for (int n = 0; n < NN; ++n) m = fmaxf(m, wlog[n * EE + e]);
    float s = 0.f;
    for (int n = 0; n < NN; ++n) {
      float ev = __expf(wlog[n * EE + e] - m);
      dw[n * EE + e] = ev; s += ev;
    }
    float inv = 1.f / s;
    for (int n = 0; n < NN; ++n) dw[n * EE + e] *= inv;
  }
  // combine softmax over E (per token row), threads 64..127
  if (t >= 64 && t < 128) {
    int n = t - 64;
    float m = -1e30f;
#pragma unroll
    for (int e = 0; e < EE; ++e) m = fmaxf(m, wlog[n * EE + e]);
    float tmp[EE];
    float s = 0.f;
#pragma unroll
    for (int e = 0; e < EE; ++e) { tmp[e] = __expf(wlog[n * EE + e] - m); s += tmp[e]; }
    float inv = 1.f / s;
#pragma unroll
    for (int e = 0; e < EE; ++e)
      cw_out[((size_t)b * NN + n) * EE + e] = tmp[e] * inv;
  }
  __syncthreads();

  // expert_in[e][d] = sum_n dw[n][e] * x[n][d]; 2560 outputs, 10/thread
  for (int j = 0; j < 10; ++j) {
    int i = j * 256 + t;                 // 0..2559
    int e = i / DDIM, d = i % DDIM;
    float acc = 0.f;
    for (int n = 0; n < NN; ++n) acc += dw[n * EE + e] * xs[n * DDIM + d];
    xin_out[((size_t)e * BB + b) * DDIM + d] = f2bf(acc);
  }
}

// ---------------------------------------------------------------------------
// Kernel 2: fused expert MLP, weight-stationary in LDS, bf16 WMMA.
// grid = (B/64 token tiles, E experts); 256 threads = 8 wave32.
// ---------------------------------------------------------------------------
__global__ __launch_bounds__(256) void k_mlp(
    const __bf16* __restrict__ xin,       // [E][B][160] bf16
    const float* __restrict__ W1, const float* __restrict__ b1,
    const float* __restrict__ W2, const float* __restrict__ b2,
    const float* __restrict__ W3, const float* __restrict__ b3,
    float* __restrict__ eo) {             // [B][E][128] f32
  extern __shared__ char smem[];
  __bf16* w1t = (__bf16*)smem;            // [256][160]  (W1^T)  80 KB
  __bf16* w2t = w1t + HH * DDIM;          // [256][256]  (W2^T) 128 KB
  __bf16* w3t = w2t + HH * HH;            // [128][256]  (W3^T)  64 KB
  __bf16* act = w3t + LDIM * HH;          // [64][256] activations 32 KB
  float*  b1s = (float*)(act + 64 * HH);
  float*  b2s = b1s + HH;
  float*  b3s = b2s + HH;

  const int t = threadIdx.x;
  const int e = blockIdx.y;
  const int btile = blockIdx.x * 64;

  // ---- stage weights transposed to [out][in], f32 -> bf16 ----
  for (int i = t; i < DDIM * HH; i += 256) {
    int d = i >> 8, h = i & 255;
    w1t[h * DDIM + d] = f2bf(W1[((size_t)e * DDIM + d) * HH + h]);
  }
  for (int i = t; i < HH * HH; i += 256) {
    int k = i >> 8, h = i & 255;
    w2t[h * HH + k] = f2bf(W2[((size_t)e * HH + k) * HH + h]);
  }
  for (int i = t; i < HH * LDIM; i += 256) {
    int k = i >> 7, d = i & 127;
    w3t[d * HH + k] = f2bf(W3[((size_t)e * HH + k) * LDIM + d]);
  }
  b1s[t] = b1[e * HH + t];
  b2s[t] = b2[e * HH + t];
  if (t < LDIM) b3s[t] = b3[e * LDIM + t];

  const int lane = t & 31;
  const int wv   = t >> 5;          // 0..7
  const int nrow = lane & 15;       // A row / B col / D col within tile
  const int half = lane >> 4;       // 0,1
  const int m0   = (wv & 3) * 16;   // token-row tile
  const int ng   = wv >> 2;         // 0..1 -> which 8 N-tiles

  // prefetch next token tile's A rows while we do LDS staging
  if (btile + 64 < BB)
    __builtin_prefetch(xin + ((size_t)e * BB + btile + 64 + m0 + nrow) * DDIM, 0, 1);

  __syncthreads();

  // ---- GEMM1: h1 = relu(X @ W1 + b1); X from global, 64x160 -> 64x256 ----
  {
    v8f acc[8] = {};
    const __bf16* abase =
        xin + ((size_t)e * BB + btile + m0 + nrow) * DDIM + half * 8;
#pragma unroll
    for (int ks = 0; ks < 5; ++ks) {
      v16bf af = ld_frag_a(abase + ks * 32);
#pragma unroll
      for (int j = 0; j < 8; ++j) {
        int nt = ng * 8 + j;
        v16bf bfr = ld_frag_a(w1t + (nt * 16 + nrow) * DDIM + ks * 32 + half * 8);
        acc[j] = __builtin_amdgcn_wmma_f32_16x16x32_bf16(
            false, af, false, bfr, (short)0, acc[j], false, false);
      }
    }
#pragma unroll
    for (int j = 0; j < 8; ++j) {
      int h = (ng * 8 + j) * 16 + nrow;
      float bias = b1s[h];
#pragma unroll
      for (int r = 0; r < 8; ++r) {
        int m = m0 + half * 8 + r;
        float v = acc[j][r] + bias;
        act[m * HH + h] = f2bf(v > 0.f ? v : 0.f);
      }
    }
  }
  __syncthreads();

  // ---- GEMM2: h2 = relu(h1 @ W2 + b2); 64x256 -> 64x256, A from LDS ----
  {
    v8f acc[8] = {};
#pragma unroll
    for (int ks = 0; ks < 8; ++ks) {
      v16bf af = ld_frag_a(act + (m0 + nrow) * HH + ks * 32 + half * 8);
#pragma unroll
      for (int j = 0; j < 8; ++j) {
        int nt = ng * 8 + j;
        v16bf bfr = ld_frag_a(w2t + (nt * 16 + nrow) * HH + ks * 32 + half * 8);
        acc[j] = __builtin_amdgcn_wmma_f32_16x16x32_bf16(
            false, af, false, bfr, (short)0, acc[j], false, false);
      }
    }
    __syncthreads();   // everyone finished reading h1 before overwrite
#pragma unroll
    for (int j = 0; j < 8; ++j) {
      int h = (ng * 8 + j) * 16 + nrow;
      float bias = b2s[h];
#pragma unroll
      for (int r = 0; r < 8; ++r) {
        int m = m0 + half * 8 + r;
        float v = acc[j][r] + bias;
        act[m * HH + h] = f2bf(v > 0.f ? v : 0.f);
      }
    }
  }
  __syncthreads();

  // ---- GEMM3: eo = h2 @ W3 + b3; 64x256 -> 64x128, write f32 global ----
  {
    v8f acc[4] = {};
#pragma unroll
    for (int ks = 0; ks < 8; ++ks) {
      v16bf af = ld_frag_a(act + (m0 + nrow) * HH + ks * 32 + half * 8);
#pragma unroll
      for (int j = 0; j < 4; ++j) {
        int nt = ng * 4 + j;
        v16bf bfr = ld_frag_a(w3t + (nt * 16 + nrow) * HH + ks * 32 + half * 8);
        acc[j] = __builtin_amdgcn_wmma_f32_16x16x32_bf16(
            false, af, false, bfr, (short)0, acc[j], false, false);
      }
    }
#pragma unroll
    for (int j = 0; j < 4; ++j) {
      int d = (ng * 4 + j) * 16 + nrow;
      float bias = b3s[d];
#pragma unroll
      for (int r = 0; r < 8; ++r) {
        int bb = btile + m0 + half * 8 + r;
        eo[((size_t)bb * EE + e) * LDIM + d] = acc[j][r] + bias;
      }
    }
  }
}

// ---------------------------------------------------------------------------
// Kernel 3: combine.  One block per batch element.
// ---------------------------------------------------------------------------
__global__ __launch_bounds__(256) void k_combine(
    const float* __restrict__ cw,    // [B][N][E]
    const float* __restrict__ eo,    // [B][E][128]
    float* __restrict__ out) {       // [B][N][128]
  __shared__ float eos[EE * LDIM];   // 8 KB
  __shared__ float cws[NN * EE];     // 4 KB
  const int b = blockIdx.x;
  const int t = threadIdx.x;
  for (int i = t; i < EE * LDIM; i += 256) eos[i] = eo[(size_t)b * EE * LDIM + i];
  for (int i = t; i < NN * EE;   i += 256) cws[i] = cw[(size_t)b * NN * EE + i];
  __syncthreads();
  for (int i = t; i < NN * LDIM; i += 256) {
    int n = i >> 7, d = i & (LDIM - 1);
    float acc = 0.f;
#pragma unroll
    for (int e = 0; e < EE; ++e) acc += cws[n * EE + e] * eos[e * LDIM + d];
    out[(size_t)b * NN * LDIM + i] = acc;
  }
}

// ---------------------------------------------------------------------------
extern "C" void kernel_launch(void* const* d_in, const int* in_sizes, int n_in,
                              void* d_out, int out_size, void* d_ws, size_t ws_size,
                              hipStream_t stream) {
  (void)in_sizes; (void)n_in; (void)out_size; (void)ws_size;
  const float* z   = (const float*)d_in[0];
  const float* a   = (const float*)d_in[1];
  const float* phi = (const float*)d_in[2];
  const float* W1  = (const float*)d_in[3];
  const float* b1  = (const float*)d_in[4];
  const float* W2  = (const float*)d_in[5];
  const float* b2  = (const float*)d_in[6];
  const float* W3  = (const float*)d_in[7];
  const float* b3  = (const float*)d_in[8];
  float* out = (float*)d_out;

  // workspace layout
  const size_t CW_BYTES  = (size_t)BB * NN * EE * sizeof(float);       // 16 MB
  const size_t XIN_BYTES = (size_t)EE * BB * DDIM * sizeof(__bf16);    // 20 MB
  char* ws = (char*)d_ws;
  float*  cw  = (float*)ws;
  __bf16* xin = (__bf16*)(ws + CW_BYTES);
  float*  eo  = (float*)(ws + CW_BYTES + XIN_BYTES);                   // 32 MB

  // 272KB bf16 weights + 32KB act + biases: needs large dynamic LDS (CDNA5 WGP = 320KB)
  const size_t lds2 = (size_t)(HH * DDIM + HH * HH + LDIM * HH + 64 * HH) * sizeof(__bf16)
                    + (size_t)(HH + HH + LDIM) * sizeof(float);        // 313856 B
  (void)hipFuncSetAttribute(reinterpret_cast<const void*>(&k_mlp),
                            hipFuncAttributeMaxDynamicSharedMemorySize, (int)lds2);

  k_route<<<dim3(BB), 256, 0, stream>>>(z, a, phi, cw, xin);
  k_mlp<<<dim3(BB / 64, EE), 256, lds2, stream>>>(xin, W1, b1, W2, b2, W3, b3, eo);
  k_combine<<<dim3(BB), 256, 0, stream>>>(cw, eo, out);
}
